// Interaction_10282151706847
// MI455X (gfx1250) — matrix operand order repaired and checked by reading, not compiled, not run on hardware
//
#include <hip/hip_runtime.h>
#include <hip/hip_bf16.h>
#include <math.h>

#define HDIM 256
#define GPAD 64

typedef __attribute__((ext_vector_type(16))) __bf16 v16bf;
typedef __attribute__((ext_vector_type(8)))  float  v8f;

union FragU {
  uint4 u4[2];
  v16bf v;
};

__device__ __forceinline__ unsigned short f2bf(float x) {
  unsigned int u = __float_as_uint(x);
  u += 0x7FFFu + ((u >> 16) & 1u);           // round-to-nearest-even
  return (unsigned short)(u >> 16);
}
__device__ __forceinline__ float bf2f(unsigned short b) {
  return __uint_as_float(((unsigned int)b) << 16);
}
// shifted softplus: softplus(x) - log(2), numerically stable
__device__ __forceinline__ float ssp_f(float x) {
  return fmaxf(x, 0.0f) + log1pf(__expf(-fabsf(x))) - 0.69314718055994531f;
}
__device__ __forceinline__ v8f vzero() {
  v8f z = {0.f, 0.f, 0.f, 0.f, 0.f, 0.f, 0.f, 0.f};
  return z;
}

// A fragment (16x32 bf16, MxK) from a row-major bf16 LDS tile.
// ISA layout: lanes 0-15 row M=lane, K = k0+0..7 (v0-3) and k0+16..23 (v4-7);
// lanes 16-31 row M=lane-16, K = k0+8..15 and k0+24..31.
__device__ __forceinline__ v16bf load_a_frag_lds(const unsigned short* tile, int strideElem,
                                                 int m0, int k0, int lane) {
  int row = m0 + (lane & 15);
  int kk  = k0 + ((lane & 16) >> 1);   // +8 for hi half-wave
  const unsigned short* p = tile + row * strideElem + kk;
  FragU f;
  f.u4[0] = *(const uint4*)(p);        // VGPR 0-3
  f.u4[1] = *(const uint4*)(p + 16);   // VGPR 4-7
  return f.v;
}

// B fragment (32x16 bf16, KxN) from pre-swizzled pack (512 u16 per fragment):
// chunk0 (v0-3) at frag*512 + lane*8, chunk1 (v4-7) at +256.
__device__ __forceinline__ v16bf load_b_frag(const unsigned short* pack, int frag, int lane) {
  const unsigned short* p = pack + (size_t)frag * 512 + lane * 8;
  FragU f;
  f.u4[0] = *(const uint4*)(p);
  f.u4[1] = *(const uint4*)(p + 256);
  return f.v;
}

// Wave computes a 32x64 tile: rows wm*32..+31 (two 16-row subtiles),
// cols wn*64..+63 (4 ntiles), K = ksteps*32.  Each B fragment feeds 2 WMMAs.
__device__ __forceinline__ void gemm_tile64(const unsigned short* Atile, int strideElem,
                                            const unsigned short* pack, int ksteps,
                                            int wm, int wn, int lane, v8f acc[8]) {
  for (int kt = 0; kt < ksteps; ++kt) {
    v16bf a0 = load_a_frag_lds(Atile, strideElem, wm * 32,      kt * 32, lane);
    v16bf a1 = load_a_frag_lds(Atile, strideElem, wm * 32 + 16, kt * 32, lane);
#pragma unroll
    for (int nt = 0; nt < 4; ++nt) {
      v16bf b = load_b_frag(pack, kt * 16 + wn * 4 + nt, lane);
      acc[nt] = __builtin_amdgcn_wmma_f32_16x16x32_bf16(
          false, a0, false, b, (short)0, acc[nt], false, false);
      acc[4 + nt] = __builtin_amdgcn_wmma_f32_16x16x32_bf16(
          false, a1, false, b, (short)0, acc[4 + nt], false, false);
    }
  }
}

__global__ void zero_kernel(float* __restrict__ p, long n) {
  long i = (long)blockIdx.x * blockDim.x + threadIdx.x;
  long stride = (long)gridDim.x * blockDim.x;
  for (; i < n; i += stride) p[i] = 0.0f;
}

// Pack W[K x Nf] (f32, row-major) into WMMA B fragments (bf16), K padded to Kpad.
__global__ void pack_b_kernel(const float* __restrict__ W, int K, int Kpad, int Nf,
                              unsigned short* __restrict__ out) {
  int t = blockIdx.x * blockDim.x + threadIdx.x;
  int ntiles = Nf / 16;
  int nFrag = (Kpad / 32) * ntiles;
  if (t >= nFrag * 32) return;
  int lane = t & 31, frag = t >> 5;
  int kt = frag / ntiles, nt = frag % ntiles;
  int n = nt * 16 + (lane & 15);
  int kbase = kt * 32 + ((lane & 16) ? 8 : 0);
  unsigned short* o = out + (size_t)frag * 512 + lane * 8;
#pragma unroll
  for (int j = 0; j < 8; ++j) {
    int k0 = kbase + j, k1 = kbase + 16 + j;
    o[j]       = f2bf(k0 < K ? W[(size_t)k0 * Nf + n] : 0.0f);
    o[j + 256] = f2bf(k1 < K ? W[(size_t)k1 * Nf + n] : 0.0f);
  }
}

// h1 = h @ aw_W + aw_b, stored bf16 (L2-resident gather table)
__global__ void __launch_bounds__(256) h1_kernel(const float* __restrict__ h,
                          const float* __restrict__ bias,
                          const unsigned short* __restrict__ packAW,
                          unsigned short* __restrict__ h1bf, int Nn) {
  __shared__ unsigned short A[64 * HDIM];
  int tid = threadIdx.x, lane = tid & 31, w = tid >> 5;
  int n0 = blockIdx.x * 64;
  for (int i = tid; i < 64 * HDIM; i += 256) {
    int r = i >> 8, c = i & 255;
    float v = (n0 + r < Nn) ? h[(size_t)(n0 + r) * HDIM + c] : 0.0f;
    A[i] = f2bf(v);
  }
  __syncthreads();
  int wm = w >> 2, wn = w & 3;
  v8f acc[8];
#pragma unroll
  for (int i = 0; i < 8; ++i) acc[i] = vzero();
  gemm_tile64(A, HDIM, packAW, HDIM / 32, wm, wn, lane, acc);

  int nlo = lane & 15;
#pragma unroll
  for (int m = 0; m < 2; ++m) {
    int rBase = wm * 32 + m * 16 + ((lane & 16) >> 1);
#pragma unroll
    for (int nt = 0; nt < 4; ++nt) {
      int col = wn * 64 + nt * 16 + nlo;
      float bv = bias[col];
#pragma unroll
      for (int v = 0; v < 8; ++v) {
        int node = n0 + rBase + v;
        if (node < Nn) h1bf[(size_t)node * HDIM + col] = f2bf(acc[m * 4 + nt][v] + bv);
      }
    }
  }
}

// Fused edge block: filter MLP (2 layers) + cutoff + gather/modulate/scatter-add
__global__ void __launch_bounds__(256) edge_kernel(const float* __restrict__ edge_attr,
                            const int* __restrict__ edge_index,
                            const float* __restrict__ edge_w,
                            const float* __restrict__ m1_b,
                            const float* __restrict__ m2_b,
                            const unsigned short* __restrict__ packM1,
                            const unsigned short* __restrict__ packM2,
                            const unsigned short* __restrict__ h1bf,
                            float* __restrict__ agg,
                            int E_, int G_) {
  __shared__ unsigned short Atile[64 * GPAD];
  __shared__ unsigned short T1[64 * HDIM];
  __shared__ int   sSrc[64];
  __shared__ int   sDst[64];
  __shared__ float sC[64];
  int tid = threadIdx.x, lane = tid & 31, w = tid >> 5;
  long e0 = (long)blockIdx.x * 64;
  if (tid < 64) {
    long e = e0 + tid;
    if (e < E_) {
      sSrc[tid] = edge_index[e];
      sDst[tid] = edge_index[(long)E_ + e];
      sC[tid] = 0.5f * (__cosf(edge_w[e] * 0.31415926535897932f) + 1.0f);
    } else {
      sSrc[tid] = 0; sDst[tid] = 0; sC[tid] = 0.0f;
    }
  }
  for (int i = tid; i < 64 * GPAD; i += 256) {
    int r = i >> 6, c = i & (GPAD - 1);
    long e = e0 + r;
    float v = (e < E_ && c < G_) ? edge_attr[e * G_ + c] : 0.0f;
    Atile[i] = f2bf(v);
  }
  __syncthreads();

  int wm = w >> 2, wn = w & 3;
  int nlo = lane & 15;

  // layer 1: [64,64] @ [64,256] -> ssp -> T1 (bf16 LDS)
  v8f acc[8];
#pragma unroll
  for (int i = 0; i < 8; ++i) acc[i] = vzero();
  gemm_tile64(Atile, GPAD, packM1, GPAD / 32, wm, wn, lane, acc);
#pragma unroll
  for (int m = 0; m < 2; ++m) {
    int rBase = wm * 32 + m * 16 + ((lane & 16) >> 1);
#pragma unroll
    for (int nt = 0; nt < 4; ++nt) {
      int col = wn * 64 + nt * 16 + nlo;
      float bv = m1_b[col];
#pragma unroll
      for (int v = 0; v < 8; ++v) {
        int row = rBase + v;
        T1[row * HDIM + col] = f2bf(ssp_f(acc[m * 4 + nt][v] + bv));
      }
    }
  }
  __syncthreads();

  // layer 2: [64,256] @ [256,256] -> ssp -> *C -> gather/scale/scatter
#pragma unroll
  for (int i = 0; i < 8; ++i) acc[i] = vzero();
  gemm_tile64(T1, HDIM, packM2, HDIM / 32, wm, wn, lane, acc);
#pragma unroll
  for (int m = 0; m < 2; ++m) {
    int rBase = wm * 32 + m * 16 + ((lane & 16) >> 1);
#pragma unroll
    for (int nt = 0; nt < 4; ++nt) {
      int col = wn * 64 + nt * 16 + nlo;
      float bv = m2_b[col];
#pragma unroll
      for (int v = 0; v < 8; ++v) {
        int row = rBase + v;
        if (e0 + row < E_) {
          float wf = ssp_f(acc[m * 4 + nt][v] + bv) * sC[row];
          float g  = bf2f(h1bf[(size_t)sSrc[row] * HDIM + col]);
          __hip_atomic_fetch_add(&agg[(size_t)sDst[row] * HDIM + col], g * wf,
                                 __ATOMIC_RELAXED, __HIP_MEMORY_SCOPE_AGENT);
        }
      }
    }
  }
}

// out = ssp(agg @ o1 + b1) @ o2 + b2
__global__ void __launch_bounds__(256) out_kernel(const float* __restrict__ agg,
                           const float* __restrict__ b1,
                           const float* __restrict__ b2,
                           const unsigned short* __restrict__ packO1,
                           const unsigned short* __restrict__ packO2,
                           float* __restrict__ out, int Nn) {
  __shared__ unsigned short A[64 * HDIM];
  __shared__ unsigned short U[64 * HDIM];
  int tid = threadIdx.x, lane = tid & 31, w = tid >> 5;
  int n0 = blockIdx.x * 64;
  for (int i = tid; i < 64 * HDIM; i += 256) {
    int r = i >> 8, c = i & 255;
    float v = (n0 + r < Nn) ? agg[(size_t)(n0 + r) * HDIM + c] : 0.0f;
    A[i] = f2bf(v);
  }
  __syncthreads();
  int wm = w >> 2, wn = w & 3;
  int nlo = lane & 15;

  v8f acc[8];
#pragma unroll
  for (int i = 0; i < 8; ++i) acc[i] = vzero();
  gemm_tile64(A, HDIM, packO1, HDIM / 32, wm, wn, lane, acc);
#pragma unroll
  for (int m = 0; m < 2; ++m) {
    int rBase = wm * 32 + m * 16 + ((lane & 16) >> 1);
#pragma unroll
    for (int nt = 0; nt < 4; ++nt) {
      int col = wn * 64 + nt * 16 + nlo;
      float bv = b1[col];
#pragma unroll
      for (int v = 0; v < 8; ++v) {
        int row = rBase + v;
        U[row * HDIM + col] = f2bf(ssp_f(acc[m * 4 + nt][v] + bv));
      }
    }
  }
  __syncthreads();

#pragma unroll
  for (int i = 0; i < 8; ++i) acc[i] = vzero();
  gemm_tile64(U, HDIM, packO2, HDIM / 32, wm, wn, lane, acc);
#pragma unroll
  for (int m = 0; m < 2; ++m) {
    int rBase = wm * 32 + m * 16 + ((lane & 16) >> 1);
#pragma unroll
    for (int nt = 0; nt < 4; ++nt) {
      int col = wn * 64 + nt * 16 + nlo;
      float bv = b2[col];
#pragma unroll
      for (int v = 0; v < 8; ++v) {
        int node = n0 + rBase + v;
        if (node < Nn) out[(size_t)node * HDIM + col] = acc[m * 4 + nt][v] + bv;
      }
    }
  }
}

extern "C" void kernel_launch(void* const* d_in, const int* in_sizes, int n_in,
                              void* d_out, int out_size, void* d_ws, size_t ws_size,
                              hipStream_t stream) {
  const float* h          = (const float*)d_in[0];
  const int*   edge_index = (const int*)d_in[1];
  const float* edge_w     = (const float*)d_in[2];
  const float* edge_attr  = (const float*)d_in[3];
  const float* aw_W = (const float*)d_in[4];
  const float* aw_b = (const float*)d_in[5];
  const float* m1_W = (const float*)d_in[6];
  const float* m1_b = (const float*)d_in[7];
  const float* m2_W = (const float*)d_in[8];
  const float* m2_b = (const float*)d_in[9];
  const float* o1_W = (const float*)d_in[10];
  const float* o1_b = (const float*)d_in[11];
  const float* o2_W = (const float*)d_in[12];
  const float* o2_b = (const float*)d_in[13];

  int N = in_sizes[0] / HDIM;
  int E = in_sizes[2];
  int G = in_sizes[3] / E;

  char* ws = (char*)d_ws;
  size_t off = 0;
  auto take = [&](size_t bytes) -> void* {
    void* p = ws + off;
    off += bytes;
    off = (off + 255) & ~(size_t)255;
    return p;
  };
  unsigned short* packAW = (unsigned short*)take((size_t)8 * 16 * 1024);
  unsigned short* packM1 = (unsigned short*)take((size_t)(GPAD / 32) * 16 * 1024);
  unsigned short* packM2 = (unsigned short*)take((size_t)8 * 16 * 1024);
  unsigned short* packO1 = (unsigned short*)take((size_t)8 * 16 * 1024);
  unsigned short* packO2 = (unsigned short*)take((size_t)8 * 16 * 1024);
  unsigned short* h1bf   = (unsigned short*)take((size_t)N * HDIM * 2);
  float*          agg    = (float*)take((size_t)N * HDIM * 4);

  // 1) zero the atomic-scatter target (must happen every call)
  zero_kernel<<<dim3(2048), dim3(256), 0, stream>>>(agg, (long)N * HDIM);

  // 2) pack weights into WMMA B-fragment layout (bf16)
  auto packBlocks = [](int Kpad) { return ((Kpad / 32) * (HDIM / 16) * 32 + 255) / 256; };
  pack_b_kernel<<<dim3(packBlocks(HDIM)), dim3(256), 0, stream>>>(aw_W, HDIM, HDIM, HDIM, packAW);
  pack_b_kernel<<<dim3(packBlocks(GPAD)), dim3(256), 0, stream>>>(m1_W, G, GPAD, HDIM, packM1);
  pack_b_kernel<<<dim3(packBlocks(HDIM)), dim3(256), 0, stream>>>(m2_W, HDIM, HDIM, HDIM, packM2);
  pack_b_kernel<<<dim3(packBlocks(HDIM)), dim3(256), 0, stream>>>(o1_W, HDIM, HDIM, HDIM, packO1);
  pack_b_kernel<<<dim3(packBlocks(HDIM)), dim3(256), 0, stream>>>(o2_W, HDIM, HDIM, HDIM, packO2);

  int nodeBlocks = (N + 63) / 64;
  int edgeBlocks = (E + 63) / 64;

  // 3) h1 = h @ aw_W + aw_b  (bf16 table)
  h1_kernel<<<dim3(nodeBlocks), dim3(256), 0, stream>>>(h, aw_b, packAW, h1bf, N);

  // 4) fused edge MLP + CFConv scatter
  edge_kernel<<<dim3(edgeBlocks), dim3(256), 0, stream>>>(
      edge_attr, edge_index, edge_w, m1_b, m2_b, packM1, packM2, h1bf, agg, E, G);

  // 5) output MLP
  out_kernel<<<dim3(nodeBlocks), dim3(256), 0, stream>>>(
      agg, o1_b, o2_b, packO1, packO2, (float*)d_out, N);
}
